// SelfAttention_6768868458553
// MI455X (gfx1250) — matrix operand order repaired and checked by reading, not compiled, not run on hardware
//
#include <hip/hip_runtime.h>

// ---------------------------------------------------------------------------
// Self-attention for MI455X (gfx1250): B=4, L=2048, DIM=768, H=12, C=64.
// Matmuls via v_wmma_f32_16x16x32_f16; all tile staging via
// global_load_async_to_lds_b128 (ASYNCcnt) with double-buffered LDS.
// ---------------------------------------------------------------------------

typedef _Float16 h4  __attribute__((ext_vector_type(4)));
typedef _Float16 h8  __attribute__((ext_vector_type(8)));
typedef _Float16 h16 __attribute__((ext_vector_type(16)));
typedef float    f4  __attribute__((ext_vector_type(4)));
typedef float    f8  __attribute__((ext_vector_type(8)));

#define BATCH 4
#define SEQ   2048
#define DIM   768
#define NH    12
#define HD    64

__device__ __forceinline__ f8 wmma_f16(h16 a, h16 b, f8 c) {
    return __builtin_amdgcn_wmma_f32_16x16x32_f16(false, a, false, b, (short)0, c,
                                                  false, false);
}

__device__ __forceinline__ h16 cat8(h8 a, h8 b) {
    return __builtin_shufflevector(a, b, 0, 1, 2, 3, 4, 5, 6, 7,
                                   8, 9, 10, 11, 12, 13, 14, 15);
}

__device__ __forceinline__ h16 ld16(const _Float16* p) {
    h8 a = *(const h8*)p;
    h8 b = *(const h8*)(p + 8);
    return cat8(a, b);
}

// Async 16-byte global -> LDS copy (per lane). GVS addressing:
// mem = SGPR base + 32-bit per-lane byte offset. LDS address = low 32 bits of
// the generic pointer into shared memory. Tracked by ASYNCcnt.
__device__ __forceinline__ void async_cp16(const _Float16* lds_dst,
                                           const _Float16* gbase,
                                           unsigned byte_off) {
    unsigned lds = (unsigned)(unsigned long long)lds_dst;
    unsigned long long base = (unsigned long long)gbase;
    asm volatile("global_load_async_to_lds_b128 %0, %1, %2"
                 :: "v"(lds), "v"(byte_off), "s"(base)
                 : "memory");
}

// ---------------------------------------------------------------------------
// Kernel 1a: fp32 -> f16 conversion of x (vectorized)
// ---------------------------------------------------------------------------
__global__ void convert_x(const float* __restrict__ x, _Float16* __restrict__ xh) {
    const int total4 = (BATCH * SEQ * DIM) / 4;
    for (int i = blockIdx.x * blockDim.x + threadIdx.x; i < total4;
         i += gridDim.x * blockDim.x) {
        f4 v = *(const f4*)(x + i * 4);
        h4 r;
        r[0] = (_Float16)v[0]; r[1] = (_Float16)v[1];
        r[2] = (_Float16)v[2]; r[3] = (_Float16)v[3];
        *(h4*)(xh + i * 4) = r;
    }
}

// ---------------------------------------------------------------------------
// Kernel 1b: weights fp32 -> f16, transposed: Wt[n][k] = W[k][n]
// ---------------------------------------------------------------------------
__global__ void transpose_weights(const float* __restrict__ w1,
                                  const float* __restrict__ w2,
                                  _Float16* __restrict__ wt1,
                                  _Float16* __restrict__ wt2) {
    const int N1 = DIM * 3 * DIM;
    const int N2 = DIM * DIM;
    for (int id = blockIdx.x * blockDim.x + threadIdx.x; id < N1 + N2;
         id += gridDim.x * blockDim.x) {
        if (id < N1) {
            int k = id / (3 * DIM);
            int n = id - k * (3 * DIM);
            wt1[(size_t)n * DIM + k] = (_Float16)w1[id];
        } else {
            int id2 = id - N1;
            int k = id2 / DIM;
            int n = id2 - k * DIM;
            wt2[(size_t)n * DIM + k] = (_Float16)w2[id2];
        }
    }
}

// ---------------------------------------------------------------------------
// Kernel 2/4: tiled WMMA GEMM, C[M,N] = A[M,768] * Bt[N,768]^T
//   mode 0: N=2304 qkv -> scatter Qh/Kh [b,h,l,c], Vt [b,h,c,l]
//   mode 1: N=768  out = acc + bias (fp32)
// ---------------------------------------------------------------------------
#define BM 128
#define BN 128
#define BK 32
#define BKP 40
#define NT (DIM / BK)

__global__ __launch_bounds__(256)
void gemm_wmma(const _Float16* __restrict__ A, const _Float16* __restrict__ Bt,
               int mode,
               _Float16* __restrict__ Qh, _Float16* __restrict__ Kh,
               _Float16* __restrict__ Vt,
               float* __restrict__ out, const float* __restrict__ bias) {
    __shared__ _Float16 sm[2][2][BM * BKP];

    const int tid   = threadIdx.x;
    const int mBase = blockIdx.y * BM;
    const int nBase = blockIdx.x * BN;
    const int lane  = tid & 31;
    const int wave  = tid >> 5;
    const int lo    = lane & 15;
    const int hi    = lane >> 4;
    const int wm    = wave & 1;
    const int wn    = wave >> 1;

    const int row0 = (tid * 2) >> 2;
    const int kc0  = ((tid * 2) & 3) << 3;
    const int row1 = (tid * 2 + 1) >> 2;
    const int kc1  = ((tid * 2 + 1) & 3) << 3;

    f8 acc[4][2] = {};

    auto issue_tile = [&](int t, int buf) {
        const int k0 = t * BK;
        async_cp16(&sm[buf][0][row0 * BKP + kc0], A,
                   (unsigned)(((mBase + row0) * DIM + k0 + kc0) * 2));
        async_cp16(&sm[buf][0][row1 * BKP + kc1], A,
                   (unsigned)(((mBase + row1) * DIM + k0 + kc1) * 2));
        async_cp16(&sm[buf][1][row0 * BKP + kc0], Bt,
                   (unsigned)(((nBase + row0) * DIM + k0 + kc0) * 2));
        async_cp16(&sm[buf][1][row1 * BKP + kc1], Bt,
                   (unsigned)(((nBase + row1) * DIM + k0 + kc1) * 2));
    };

    issue_tile(0, 0);

    for (int t = 0; t < NT; ++t) {
        const int buf = t & 1;
        if (t + 1 < NT) {
            issue_tile(t + 1, buf ^ 1);
            asm volatile("s_wait_asynccnt 0x4" ::: "memory");
        } else {
            asm volatile("s_wait_asynccnt 0x0" ::: "memory");
        }
        __syncthreads();

        const _Float16* lA = sm[buf][0];
        const _Float16* lB = sm[buf][1];

        h16 af[4];
#pragma unroll
        for (int sm_ = 0; sm_ < 4; ++sm_) {
            const _Float16* p = lA + (wm * 64 + sm_ * 16 + lo) * BKP + hi * 8;
            af[sm_] = cat8(*(const h8*)p, *(const h8*)(p + 16));
        }
        h16 bf[2];
#pragma unroll
        for (int sn = 0; sn < 2; ++sn) {
            const _Float16* p = lB + (wn * 32 + sn * 16 + lo) * BKP + hi * 8;
            bf[sn] = cat8(*(const h8*)p, *(const h8*)(p + 16));
        }
#pragma unroll
        for (int sm_ = 0; sm_ < 4; ++sm_)
#pragma unroll
            for (int sn = 0; sn < 2; ++sn)
                acc[sm_][sn] = wmma_f16(af[sm_], bf[sn], acc[sm_][sn]);

        __syncthreads();
    }

    if (mode == 0) {
#pragma unroll
        for (int sn = 0; sn < 2; ++sn) {
            int gc  = nBase + wn * 32 + sn * 16 + lo;
            int s   = gc / DIM;
            int rem = gc - s * DIM;
            int hh  = rem >> 6;
            int cc  = rem & 63;
#pragma unroll
            for (int sm_ = 0; sm_ < 4; ++sm_) {
#pragma unroll
                for (int i = 0; i < 8; ++i) {
                    int gr = mBase + wm * 64 + sm_ * 16 + hi * 8 + i;
                    int bb = gr >> 11;
                    int ll = gr & 2047;
                    _Float16 v = (_Float16)acc[sm_][sn][i];
                    size_t bh = (size_t)(bb * NH + hh);
                    if (s == 0)
                        Qh[(bh * SEQ + ll) * HD + cc] = v;
                    else if (s == 1)
                        Kh[(bh * SEQ + ll) * HD + cc] = v;
                    else
                        Vt[(bh * HD + cc) * SEQ + ll] = v;
                }
            }
        }
    } else {
#pragma unroll
        for (int sn = 0; sn < 2; ++sn) {
            int gc = nBase + wn * 32 + sn * 16 + lo;
            float bv = bias[gc];
#pragma unroll
            for (int sm_ = 0; sm_ < 4; ++sm_) {
#pragma unroll
                for (int i = 0; i < 8; ++i) {
                    int gr = mBase + wm * 64 + sm_ * 16 + hi * 8 + i;
                    out[(size_t)gr * DIM + gc] = acc[sm_][sn][i] + bv;
                }
            }
        }
    }
}

// ---------------------------------------------------------------------------
// Kernel 3: flash attention, block-cooperative K/V staging.
// Block = 4 waves, one (b,h); each wave owns 2 query tiles (32 q), so a block
// covers 128 queries. Per 32-key chunk the block async-stages K (32x64) and
// V^T (64x32) into double-buffered LDS; all 8 q-tiles consume it.
// ---------------------------------------------------------------------------
#define KP 72   // padded K-chunk LDS row (halves): [key][c]
#define VP 40   // padded V-chunk LDS row (halves): [c][key]
#define NKC (SEQ / 32)   // 64 key chunks

__global__ __launch_bounds__(128)
void attn_flash(const _Float16* __restrict__ Qh, const _Float16* __restrict__ Kh,
                const _Float16* __restrict__ Vt, _Float16* __restrict__ attn) {
    __shared__ _Float16 sK[2][32 * KP];
    __shared__ _Float16 sV[2][64 * VP];
    __shared__ _Float16 sP[4][16 * 40];

    const int tid  = threadIdx.x;
    const int wave = tid >> 5;
    const int lane = tid & 31;
    const int lo   = lane & 15;
    const int hi   = lane >> 4;
    _Float16* P = sP[wave];

    const int bh = blockIdx.x >> 4;          // 0..47
    const int qg = blockIdx.x & 15;          // 128-query group
    const int b  = bh / NH;
    const int h  = bh - b * NH;

    const _Float16* Kb = Kh + (size_t)bh * SEQ * HD;
    const _Float16* Vb = Vt + (size_t)bh * HD * SEQ;

    // Staging map: 512 x 16B per chunk, 4 per thread.
    //   e in [0,256): K  -> row = e/8  (0..31 keys), c8 = (e%8)*8
    //   e in [256,512): V -> c  = e'/4 (0..63),      k8 = (e'%4)*8
    auto issue_chunk = [&](int kc, int buf) {
        const int kt = kc * 32;
#pragma unroll
        for (int j = 0; j < 2; ++j) {
            int e   = tid * 2 + j;           // 0..255 -> K
            int row = e >> 3;
            int c8  = (e & 7) << 3;
            async_cp16(&sK[buf][row * KP + c8], Kb,
                       (unsigned)(((kt + row) * HD + c8) * 2));
            int e2 = e;                      // 0..255 -> V
            int c  = e2 >> 2;
            int k8 = (e2 & 3) << 3;
            async_cp16(&sV[buf][c * VP + k8], Vb,
                       (unsigned)((c * SEQ + kt + k8) * 2));
        }
    };

    // Q A-fragments for this wave's two 16-query tiles
    const int qrow0 = qg * 128 + wave * 32;
    h16 qa[2][2];
#pragma unroll
    for (int qs = 0; qs < 2; ++qs)
#pragma unroll
        for (int f = 0; f < 2; ++f) {
            const _Float16* p = Qh +
                ((size_t)bh * SEQ + qrow0 + qs * 16 + lo) * HD + f * 32 + hi * 8;
            qa[qs][f] = cat8(*(const h8*)p, *(const h8*)(p + 16));
        }

    float m[2][8], lsum[2][8];
    f8 o[2][4] = {};
#pragma unroll
    for (int qs = 0; qs < 2; ++qs)
#pragma unroll
        for (int i = 0; i < 8; ++i) { m[qs][i] = -1.0e30f; lsum[qs][i] = 0.0f; }

    issue_chunk(0, 0);

    for (int kc = 0; kc < NKC; ++kc) {
        const int buf = kc & 1;
        if (kc + 1 < NKC) {
            issue_chunk(kc + 1, buf ^ 1);
            asm volatile("s_wait_asynccnt 0x4" ::: "memory");
        } else {
            asm volatile("s_wait_asynccnt 0x0" ::: "memory");
        }
        __syncthreads();

        // K B-fragments (shared by both q-tiles): [t = key subtile][f = c half]
        h16 kb[2][2];
#pragma unroll
        for (int t = 0; t < 2; ++t)
#pragma unroll
            for (int f = 0; f < 2; ++f) {
                const _Float16* p = sK[buf] + (t * 16 + lo) * KP + f * 32 + hi * 8;
                kb[t][f] = cat8(*(const h8*)p, *(const h8*)(p + 16));
            }

#pragma unroll
        for (int qs = 0; qs < 2; ++qs) {
            f8 s0 = {}, s1 = {};
#pragma unroll
            for (int f = 0; f < 2; ++f) {
                s0 = wmma_f16(qa[qs][f], kb[0][f], s0);
                s1 = wmma_f16(qa[qs][f], kb[1][f], s1);
            }

            // online softmax: row r = i + hi*8, cols = lo and 16+lo
#pragma unroll
            for (int i = 0; i < 8; ++i) {
                float lm = fmaxf(s0[i], s1[i]);
                lm = fmaxf(lm, __shfl_xor(lm, 1, 32));
                lm = fmaxf(lm, __shfl_xor(lm, 2, 32));
                lm = fmaxf(lm, __shfl_xor(lm, 4, 32));
                lm = fmaxf(lm, __shfl_xor(lm, 8, 32));
                float mn = fmaxf(m[qs][i], lm);
                float sc = __expf(m[qs][i] - mn);
                m[qs][i] = mn;
                float p0 = __expf(s0[i] - mn);
                float p1 = __expf(s1[i] - mn);
                float rs = p0 + p1;
                rs += __shfl_xor(rs, 1, 32);
                rs += __shfl_xor(rs, 2, 32);
                rs += __shfl_xor(rs, 4, 32);
                rs += __shfl_xor(rs, 8, 32);
                lsum[qs][i] = lsum[qs][i] * sc + rs;
#pragma unroll
                for (int cn = 0; cn < 4; ++cn) o[qs][cn][i] *= sc;
                P[(hi * 8 + i) * 40 + lo]      = (_Float16)p0;
                P[(hi * 8 + i) * 40 + 16 + lo] = (_Float16)p1;
            }

            // LDS ops are in-order within a wave; compiler fence only.
            asm volatile("" ::: "memory");
            const _Float16* pp = P + lo * 40 + hi * 8;
            h16 pa = cat8(*(const h8*)pp, *(const h8*)(pp + 16));
            asm volatile("" ::: "memory");

#pragma unroll
            for (int cn = 0; cn < 4; ++cn) {
                const _Float16* vp = sV[buf] + (cn * 16 + lo) * VP + hi * 8;
                h16 vb = cat8(*(const h8*)vp, *(const h8*)(vp + 16));
                o[qs][cn] = wmma_f16(pa, vb, o[qs][cn]);
            }
        }
        __syncthreads();
    }

#pragma unroll
    for (int qs = 0; qs < 2; ++qs) {
        const int qbase = qrow0 + qs * 16 + hi * 8;
#pragma unroll
        for (int cn = 0; cn < 4; ++cn) {
            int c = cn * 16 + lo;
#pragma unroll
            for (int i = 0; i < 8; ++i) {
                float v = o[qs][cn][i] / lsum[qs][i];
                attn[((size_t)b * SEQ + qbase + i) * DIM + h * HD + c] =
                    (_Float16)v;
            }
        }
    }
}

// ---------------------------------------------------------------------------
// Launch
// ---------------------------------------------------------------------------
extern "C" void kernel_launch(void* const* d_in, const int* in_sizes, int n_in,
                              void* d_out, int out_size, void* d_ws, size_t ws_size,
                              hipStream_t stream) {
    const float* x    = (const float*)d_in[0];
    const float* Wqkv = (const float*)d_in[1];
    const float* Wo   = (const float*)d_in[2];
    const float* bo   = (const float*)d_in[3];
    float* out = (float*)d_out;

    char* ws = (char*)d_ws;
    size_t off = 0;
    auto carve = [&](size_t bytes) -> _Float16* {
        _Float16* p = (_Float16*)(ws + off);
        off += (bytes + 255) & ~(size_t)255;
        return p;
    };
    const size_t NX = (size_t)BATCH * SEQ * DIM;
    _Float16* xh  = carve(NX * 2);
    _Float16* wt1 = carve((size_t)3 * DIM * DIM * 2);
    _Float16* wt2 = carve((size_t)DIM * DIM * 2);
    _Float16* qh  = carve(NX * 2);
    _Float16* kh  = carve(NX * 2);
    _Float16* vt  = carve(NX * 2);
    _Float16* ah  = carve(NX * 2);

    convert_x<<<dim3(1536), 256, 0, stream>>>(x, xh);
    transpose_weights<<<dim3(2304), 256, 0, stream>>>(Wqkv, Wo, wt1, wt2);

    gemm_wmma<<<dim3((3 * DIM) / BN, (BATCH * SEQ) / BM), 256, 0, stream>>>(
        xh, wt1, 0, qh, kh, vt, nullptr, nullptr);

    attn_flash<<<dim3(BATCH * NH * (SEQ / 128)), 128, 0, stream>>>(
        qh, kh, vt, ah);

    gemm_wmma<<<dim3(DIM / BN, (BATCH * SEQ) / BM), 256, 0, stream>>>(
        ah, wt2, 1, nullptr, nullptr, nullptr, out, bo);
}
